// Decoder_38259568673422
// MI455X (gfx1250) — matrix operand order repaired and checked by reading, not compiled, not run on hardware
//
#include <hip/hip_runtime.h>
#include <hip/hip_bf16.h>
#include <math.h>

// ---------------------------------------------------------------------------
// GRU greedy decoder for MI455X (gfx1250), 3 kernels per step.
// H=1024, V=32000, STEPS=375.
//  * one-time fp32->bf16 conversion of W_ih, W_hh, lin_W into d_ws
//    (78 MB -> resident in 192 MB L2; halves bytes/step vs fp32)
//  * per step (stream-ordered, graph-capturable):
//      K_A: gi = W_ih @ x                  (WMMA GEMV, 24 blocks)
//      K_B: gates (redundant per block) -> h_new bf16 in LDS,
//           logits = lin_W @ h_new  AND  gh(s+1) = W_hh @ h_new
//           (WMMA GEMV with B from LDS, 274 blocks)
//      K_C: log-softmax + argmax + stage emb[tok] bf16   (1 block)
// ---------------------------------------------------------------------------

#define H 1024
#define V 32000
#define STEPS 375

typedef __attribute__((ext_vector_type(8)))  __bf16 v8bf;
typedef __attribute__((ext_vector_type(16))) __bf16 v16bf;
typedef __attribute__((ext_vector_type(8)))  float  v8f;

static __device__ __forceinline__ unsigned short f2bf(float f) {
    unsigned int u = __float_as_uint(f);
    u += 0x7fffu + ((u >> 16) & 1u);        // round-to-nearest-even
    return (unsigned short)(u >> 16);
}

static __device__ __forceinline__ float sigmoidf_fast(float x) {
    return 1.0f / (1.0f + __expf(-x));
}

// ---------------- fp32 -> bf16 weight conversion (once per launch) ----------
__global__ void cvt_bf16_kernel(const float* __restrict__ src,
                                unsigned short* __restrict__ dst, int n) {
    for (int i = blockIdx.x * blockDim.x + threadIdx.x; i < n;
         i += gridDim.x * blockDim.x)
        dst[i] = f2bf(src[i]);
}

// ---------------- initial state: h0 = encoder_hidden, x = emb[0] ------------
__global__ void init_kernel(const float* __restrict__ enc,
                            const float* __restrict__ emb,
                            float* __restrict__ h0,
                            unsigned short* __restrict__ h0bf,
                            unsigned short* __restrict__ xbf,
                            int* __restrict__ tok) {
    int i = threadIdx.x;                      // 1024 threads
    float hv = enc[i];
    h0[i]   = hv;
    h0bf[i] = f2bf(hv);
    xbf[i]  = f2bf(emb[i]);                   // emb[token=0]
    if (i == 0) *tok = 0;
}

// ---------------- standalone WMMA GEMV: out = W(bf16) @ vec + bias ----------
// One wave = 16 output rows; B operand = vec broadcast into all 16 columns.
__global__ void __launch_bounds__(256)
gemv_bf16_wmma_kernel(const unsigned short* __restrict__ W,
                      const unsigned short* __restrict__ vec,
                      const float* __restrict__ bias,
                      float* __restrict__ out, int rows) {
    int wave    = (blockIdx.x * blockDim.x + threadIdx.x) >> 5;
    int lane    = threadIdx.x & 31;
    int rowbase = wave * 16;
    if (rowbase >= rows) return;              // whole-wave uniform

    int half = lane >> 4;
    int m    = lane & 15;
    const unsigned short* wrow = W + (size_t)(rowbase + m) * H + half * 16;
    const unsigned short* vp   = vec + half * 16;

    v8f c = {};
#pragma unroll 4
    for (int k0 = 0; k0 < H; k0 += 32) {
        v16bf a  = *(const v16bf*)(wrow + k0);
        v16bf b  = *(const v16bf*)(vp + k0);
        c = __builtin_amdgcn_wmma_f32_16x16x32_bf16(
                false, a, false, b, (short)0, c, false, false);
    }
    if (m == 0) {
        int r0 = rowbase + half * 8;
#pragma unroll
        for (int r = 0; r < 8; ++r)
            out[r0 + r] = c[r] + bias[r0 + r];
    }
}

// ---------------- fused gates + logits GEMV + next-step gh GEMV -------------
// Grid: 274 blocks x 256 threads = 2192 waves.
//   waves    0..1999 : logits rows  (2000 x 16 = 32000)
//   waves 2000..2191 : gh_out rows  ( 192 x 16 =  3072)
// Every block redundantly computes the full gate math so h_new (bf16) is
// available in LDS as the WMMA B operand without an extra kernel boundary.
__global__ void __launch_bounds__(256)
fused_step_kernel(const float* __restrict__ gi,        // [3H] = W_ih@x + b_ih
                  const float* __restrict__ gh,        // [3H] = W_hh@h + b_hh
                  const float* __restrict__ h_in,      // [H]
                  const unsigned short* __restrict__ Whhbf,
                  const float* __restrict__ bhh,
                  float* __restrict__ gh_out,          // [3H] for step s+1
                  const unsigned short* __restrict__ linWbf,
                  const float* __restrict__ linb,
                  float* __restrict__ logits,          // [V]
                  float* __restrict__ h_out,           // [H] ping-pong
                  float* __restrict__ hfin, int is_last) {
    __shared__ __align__(64) unsigned short hnew_bf[H];
    int t = threadIdx.x;

    // --- GRU gate math: 4 elements per thread (1024 total) ---
#pragma unroll
    for (int i = 0; i < 4; ++i) {
        int j = t * 4 + i;
        float r  = sigmoidf_fast(gi[j]         + gh[j]);
        float z  = sigmoidf_fast(gi[H + j]     + gh[H + j]);
        float n  = tanhf(gi[2 * H + j] + r * gh[2 * H + j]);
        float hn = (1.0f - z) * n + z * h_in[j];
        hnew_bf[j] = f2bf(hn);
        if (blockIdx.x == 0) {
            h_out[j] = hn;
            if (is_last) hfin[j] = hn;
        }
    }
    __syncthreads();

    // --- WMMA GEMV tiles, B operand from LDS ---
    int gw   = blockIdx.x * 8 + (t >> 5);
    int lane = t & 31;
    int half = lane >> 4;
    int m    = lane & 15;

    const unsigned short* W;
    const float* bias;
    float* out;
    int rowbase;
    if (gw < 2000) {                      // wave-uniform branch
        W = linWbf;  bias = linb; out = logits;  rowbase = gw * 16;
    } else {
        W = Whhbf;   bias = bhh;  out = gh_out;  rowbase = (gw - 2000) * 16;
    }
    const unsigned short* wrow = W + (size_t)(rowbase + m) * H + half * 16;
    const unsigned short* vp   = hnew_bf + half * 16;   // LDS

    v8f c = {};
#pragma unroll 4
    for (int k0 = 0; k0 < H; k0 += 32) {
        v16bf a  = *(const v16bf*)(wrow + k0);
        v16bf b  = *(const v16bf*)(vp + k0);            // ds_load
        c = __builtin_amdgcn_wmma_f32_16x16x32_bf16(
                false, a, false, b, (short)0, c, false, false);
    }
    if (m == 0) {
        int r0 = rowbase + half * 8;
#pragma unroll
        for (int r = 0; r < 8; ++r)
            out[r0 + r] = c[r] + bias[r0 + r];
    }
}

// ---------------- log-softmax + argmax + next-embedding prep ----------------
__global__ void __launch_bounds__(1024)
softmax_argmax_kernel(const float* __restrict__ logits,
                      const float* __restrict__ emb,
                      float* __restrict__ outrow,
                      unsigned short* __restrict__ xbf,
                      int* __restrict__ tok) {
    __shared__ float smax[1024];
    __shared__ int   sidx[1024];
    __shared__ float ssum[1024];
    int t = threadIdx.x;

    float bm = -INFINITY; int bi = 0x7fffffff;
    for (int i = t; i < V; i += 1024) {
        float v = logits[i];
        if (v > bm || (v == bm && i < bi)) { bm = v; bi = i; }
    }
    smax[t] = bm; sidx[t] = bi; __syncthreads();
    for (int s = 512; s > 0; s >>= 1) {
        if (t < s) {
            float o = smax[t + s]; int oi = sidx[t + s];
            if (o > smax[t] || (o == smax[t] && oi < sidx[t])) {
                smax[t] = o; sidx[t] = oi;
            }
        }
        __syncthreads();
    }
    float m  = smax[0];
    int   am = sidx[0];

    float ls = 0.0f;
    for (int i = t; i < V; i += 1024) ls += __expf(logits[i] - m);
    ssum[t] = ls; __syncthreads();
    for (int s = 512; s > 0; s >>= 1) {
        if (t < s) ssum[t] += ssum[t + s];
        __syncthreads();
    }
    float lse = m + __logf(ssum[0]);

    for (int i = t; i < V; i += 1024) outrow[i] = logits[i] - lse;
    if (t == 0) *tok = am;
    // stage next-step embedding in bf16 (am is uniform via shared memory)
    xbf[t] = f2bf(emb[(size_t)am * H + t]);
}

// ---------------------------------------------------------------------------
extern "C" void kernel_launch(void* const* d_in, const int* in_sizes, int n_in,
                              void* d_out, int out_size, void* d_ws,
                              size_t ws_size, hipStream_t stream) {
    const float* enc  = (const float*)d_in[0];   // [1,1,H]
    const float* emb  = (const float*)d_in[1];   // [V,H]
    const float* Wih  = (const float*)d_in[2];   // [3H,H]
    const float* Whh  = (const float*)d_in[3];   // [3H,H]
    const float* bih  = (const float*)d_in[4];   // [3H]
    const float* bhh  = (const float*)d_in[5];   // [3H]
    const float* linW = (const float*)d_in[6];   // [V,H]
    const float* linb = (const float*)d_in[7];   // [V]
    float* out = (float*)d_out;                  // [STEPS*V] ++ [H]

    // ---- workspace layout (bytes), ~78.3 MB, 256B-aligned offsets ----
    char* ws = (char*)d_ws;
    unsigned short* linWbf = (unsigned short*)(ws);                // 65,536,000
    unsigned short* Wihbf  = (unsigned short*)(ws + 65536000);     //  6,291,456
    unsigned short* Whhbf  = (unsigned short*)(ws + 71827456);     //  6,291,456
    float*          gibuf  = (float*)(ws + 78118912);              // [3H]
    float*          ghbuf0 = (float*)(ws + 78131200);              // [3H]
    float*          ghbuf1 = (float*)(ws + 78143488);              // [3H]
    float*          hbuf0  = (float*)(ws + 78155776);              // [H]
    float*          hbuf1  = (float*)(ws + 78159872);              // [H]
    unsigned short* xbf    = (unsigned short*)(ws + 78163968);     // [H] bf16
    unsigned short* h0bf   = (unsigned short*)(ws + 78166016);     // [H] bf16
    float*          logit  = (float*)(ws + 78168064);              // [V]
    int*            tok    = (int*)(ws + 78296064);

    float* hbufs[2]  = { hbuf0, hbuf1 };
    float* ghbufs[2] = { ghbuf0, ghbuf1 };

    // one-time conversions + bootstrap (every launch; deterministic)
    cvt_bf16_kernel<<<2048, 256, 0, stream>>>(linW, linWbf, V * H);
    cvt_bf16_kernel<<<1024, 256, 0, stream>>>(Wih, Wihbf, 3 * H * H);
    cvt_bf16_kernel<<<1024, 256, 0, stream>>>(Whh, Whhbf, 3 * H * H);
    init_kernel<<<1, 1024, 0, stream>>>(enc, emb, hbuf0, h0bf, xbf, tok);
    gemv_bf16_wmma_kernel<<<24, 256, 0, stream>>>(   // gh(0) = W_hh@h0 + b_hh
        Whhbf, h0bf, bhh, ghbuf0, 3 * H);

    for (int s = 0; s < STEPS; ++s) {
        int p = s & 1;
        // K_A: gi = W_ih @ x + b_ih   (x staged by previous K_C / init)
        gemv_bf16_wmma_kernel<<<24, 256, 0, stream>>>(
            Wihbf, xbf, bih, gibuf, 3 * H);
        // K_B: gates -> h_new; logits = lin_W@h_new; gh(s+1) = W_hh@h_new
        fused_step_kernel<<<274, 256, 0, stream>>>(
            gibuf, ghbufs[p], hbufs[p],
            Whhbf, bhh, ghbufs[1 - p],
            linWbf, linb, logit,
            hbufs[1 - p], out + (size_t)STEPS * V,
            (s == STEPS - 1) ? 1 : 0);
        // K_C: log-softmax row + greedy token + stage next embedding
        softmax_argmax_kernel<<<1, 1024, 0, stream>>>(
            logit, emb, out + (size_t)s * V, xbf, tok);
    }
}
